// QuantizedLlamaMLP_75368086110396
// MI455X (gfx1250) — 8-bit matrix operand order repaired and checked by reading, compiled, not run on hardware
//
#include <hip/hip_runtime.h>

typedef int v8i __attribute__((ext_vector_type(8)));
typedef int v4i_g __attribute__((vector_size(16)));
typedef __attribute__((address_space(1))) v4i_g as1_v4i;
typedef __attribute__((address_space(3))) v4i_g as3_v4i;

#define T_TOK 8192
#define H_DIM 4096
#define I_DIM 11008

#define PANEL      960
#define LDS_STRIDE 976
#define MSUB 4
#define NSUB 2

#define HAS_ASYNC_LDS __has_builtin(__builtin_amdgcn_global_load_async_to_lds_b128)

__global__ __launch_bounds__(256)
void quant_x_kernel(const float* __restrict__ x,
                    const float* __restrict__ scale_p,
                    int* __restrict__ qx, int n4) {
    int i = blockIdx.x * blockDim.x + threadIdx.x;
    if (i >= n4) return;
    const float inv = 1.0f / scale_p[0];
    const float4 v = ((const float4*)x)[i];
    int a = __float2int_rn(v.x * inv); a = max(-127, min(127, a));
    int b = __float2int_rn(v.y * inv); b = max(-127, min(127, b));
    int c = __float2int_rn(v.z * inv); c = max(-127, min(127, c));
    int d = __float2int_rn(v.w * inv); d = max(-127, min(127, d));
    qx[i] = (a & 0xff) | ((b & 0xff) << 8) | ((c & 0xff) << 16) | ((d & 0xff) << 24);
}

__global__ __launch_bounds__(256)
void pack_w_kernel(const int* __restrict__ w, int* __restrict__ qw, int n4) {
    int i = blockIdx.x * blockDim.x + threadIdx.x;
    if (i >= n4) return;
    const int4 v = ((const int4*)w)[i];
    qw[i] = (v.x & 0xff) | ((v.y & 0xff) << 8) | ((v.z & 0xff) << 16) | ((v.w & 0xff) << 24);
}

__device__ __forceinline__ void stage_a_panel(signed char* __restrict__ As,
                                              const signed char* __restrict__ Abase,
                                              int K, int m0, int k0, int pw, int tid) {
    const int rowc    = tid >> 2;
    const int quarter = tid & 3;
    const int bytes   = pw >> 2;
    const signed char* gsrc = Abase + (size_t)(m0 + rowc) * K + k0 + quarter * bytes;
    signed char* ldst       = As + rowc * LDS_STRIDE + quarter * bytes;
    for (int j = 0; j < bytes; j += 16) {
#if HAS_ASYNC_LDS
        __builtin_amdgcn_global_load_async_to_lds_b128(
            (as1_v4i*)(unsigned long long)(gsrc + j),
            (as3_v4i*)(unsigned)(unsigned long long)(ldst + j),
            0, 0);
#else
        *(int4*)(ldst + j) = *(const int4*)(gsrc + j);
#endif
    }
#if HAS_ASYNC_LDS
#if __has_builtin(__builtin_amdgcn_s_wait_asynccnt)
    __builtin_amdgcn_s_wait_asynccnt(0);
#else
    asm volatile("s_wait_asynccnt 0x0" ::: "memory");
#endif
#endif
}

__device__ __forceinline__ v8i load_a_frag_lds(const signed char* __restrict__ As,
                                               int tile_row, int kk, int lane) {
    const int* p = (const int*)(As + (tile_row) * LDS_STRIDE + kk + ((lane >> 4) << 3));
    v8i a;
    a[0] = p[0];  a[1] = p[1];
    a[2] = p[4];  a[3] = p[5];
    a[4] = p[8];  a[5] = p[9];
    a[6] = p[12]; a[7] = p[13];
    return a;
}

__device__ __forceinline__ v8i load_b_frag(const signed char* __restrict__ base,
                                           int stride, int col, int k, int lane) {
    const int* p = (const int*)(base + (size_t)col * stride + k + ((lane >> 4) << 3));
    v8i b;
    b[0] = p[0];  b[1] = p[1];  b[2] = p[4];  b[3] = p[5];
    b[4] = p[8];  b[5] = p[9];  b[6] = p[12]; b[7] = p[13];
    return b;
}


__global__ __launch_bounds__(256)
void gateup_kernel(const signed char* __restrict__ qx,
                   const signed char* __restrict__ wg,
                   const signed char* __restrict__ wu,
                   const float* __restrict__ ws_gate, const float* __restrict__ ws_up,
                   const float* __restrict__ b_gate,  const float* __restrict__ b_up,
                   const float* __restrict__ a_sg_p,  const float* __restrict__ a_sd_p,
                   signed char* __restrict__ qh) {
    __shared__ signed char As[64 * LDS_STRIDE];

    const int lane = threadIdx.x & 31;
    const int wave = threadIdx.x >> 5;
    const int m0 = blockIdx.y * (MSUB * 16);
    const int n0 = blockIdx.x * (8 * NSUB * 16) + wave * (NSUB * 16);

    const v8i zero = {0, 0, 0, 0, 0, 0, 0, 0};
    v8i accg[MSUB][NSUB], accu[MSUB][NSUB];
#pragma unroll
    for (int mi = 0; mi < MSUB; ++mi)
#pragma unroll
        for (int ni = 0; ni < NSUB; ++ni) { accg[mi][ni] = zero; accu[mi][ni] = zero; }

    for (int k0 = 0; k0 < H_DIM; k0 += PANEL) {
        const int pw = (H_DIM - k0 < PANEL) ? (H_DIM - k0) : PANEL;
        __syncthreads();
        stage_a_panel(As, qx, H_DIM, m0, k0, pw, threadIdx.x);
        __syncthreads();

        for (int kk = 0; kk < pw; kk += 64) {
            const int gk = k0 + kk;
            if (gk + 64 < H_DIM) {
                __builtin_prefetch(wg + (size_t)(n0 + (lane & 15)) * H_DIM + gk + 64, 0, 0);
                __builtin_prefetch(wu + (size_t)(n0 + (lane & 15)) * H_DIM + gk + 64, 0, 0);
            }
            v8i a[MSUB];
#pragma unroll
            for (int mi = 0; mi < MSUB; ++mi)
                a[mi] = load_a_frag_lds(As, mi * 16 + (lane & 15), kk, lane);
#pragma unroll
            for (int ni = 0; ni < NSUB; ++ni) {
                const int col = n0 + ni * 16 + (lane & 15);
                const v8i bg = load_b_frag(wg, H_DIM, col, gk, lane);
                const v8i bu = load_b_frag(wu, H_DIM, col, gk, lane);
#pragma unroll
                for (int mi = 0; mi < MSUB; ++mi) {
                    accg[mi][ni] = __builtin_amdgcn_wmma_i32_16x16x64_iu8(
                        true, a[mi], true, bg, accg[mi][ni], false, false);
                    accu[mi][ni] = __builtin_amdgcn_wmma_i32_16x16x64_iu8(
                        true, a[mi], true, bu, accu[mi][ni], false, false);
                }
            }
        }
    }

    const float a_sg = a_sg_p[0];
    const float inv_sd = 1.0f / a_sd_p[0];
#pragma unroll
    for (int ni = 0; ni < NSUB; ++ni) {
        const int col = n0 + ni * 16 + (lane & 15);
        const float sg = a_sg * ws_gate[col];
        const float su = a_sg * ws_up[col];
        const float bgv = b_gate[col];
        const float buv = b_up[col];
#pragma unroll
        for (int mi = 0; mi < MSUB; ++mi) {
            const int rowbase = m0 + mi * 16 + ((lane >> 4) << 3);
#pragma unroll
            for (int r = 0; r < 8; ++r) {
                const float g = (float)accg[mi][ni][r] * sg + bgv;
                const float u = (float)accu[mi][ni][r] * su + buv;
                const float h = (g / (1.0f + __expf(-g))) * u;
                int q = __float2int_rn(h * inv_sd);
                q = max(-127, min(127, q));
                qh[(size_t)(rowbase + r) * I_DIM + col] = (signed char)q;
            }
        }
    }
}

__global__ __launch_bounds__(256)
void down_kernel(const signed char* __restrict__ qh,
                 const signed char* __restrict__ wd,
                 const float* __restrict__ ws_down, const float* __restrict__ b_down,
                 const float* __restrict__ a_sd_p,
                 float* __restrict__ out) {
    __shared__ signed char As[64 * LDS_STRIDE];

    const int lane = threadIdx.x & 31;
    const int wave = threadIdx.x >> 5;
    const int m0 = blockIdx.y * (MSUB * 16);
    const int n0 = blockIdx.x * (8 * NSUB * 16) + wave * (NSUB * 16);

    const v8i zero = {0, 0, 0, 0, 0, 0, 0, 0};
    v8i acc[MSUB][NSUB];
#pragma unroll
    for (int mi = 0; mi < MSUB; ++mi)
#pragma unroll
        for (int ni = 0; ni < NSUB; ++ni) acc[mi][ni] = zero;

    for (int k0 = 0; k0 < I_DIM; k0 += PANEL) {
        const int pw = (I_DIM - k0 < PANEL) ? (I_DIM - k0) : PANEL;
        __syncthreads();
        stage_a_panel(As, qh, I_DIM, m0, k0, pw, threadIdx.x);
        __syncthreads();

        for (int kk = 0; kk < pw; kk += 64) {
            const int gk = k0 + kk;
            if (gk + 64 < I_DIM) {
                __builtin_prefetch(wd + (size_t)(n0 + (lane & 15)) * I_DIM + gk + 64, 0, 0);
            }
            v8i a[MSUB];
#pragma unroll
            for (int mi = 0; mi < MSUB; ++mi)
                a[mi] = load_a_frag_lds(As, mi * 16 + (lane & 15), kk, lane);
#pragma unroll
            for (int ni = 0; ni < NSUB; ++ni) {
                const v8i b = load_b_frag(wd, I_DIM, n0 + ni * 16 + (lane & 15), gk, lane);
#pragma unroll
                for (int mi = 0; mi < MSUB; ++mi) {
                    acc[mi][ni] = __builtin_amdgcn_wmma_i32_16x16x64_iu8(
                        true, a[mi], true, b, acc[mi][ni], false, false);
                }
            }
        }
    }

    const float a_sd = a_sd_p[0];
#pragma unroll
    for (int ni = 0; ni < NSUB; ++ni) {
        const int col = n0 + ni * 16 + (lane & 15);
        const float s = a_sd * ws_down[col];
        const float b = b_down[col];
#pragma unroll
        for (int mi = 0; mi < MSUB; ++mi) {
            const int rowbase = m0 + mi * 16 + ((lane >> 4) << 3);
#pragma unroll
            for (int r = 0; r < 8; ++r) {
                out[(size_t)(rowbase + r) * H_DIM + col] = (float)acc[mi][ni][r] * s + b;
            }
        }
    }
}

extern "C" void kernel_launch(void* const* d_in, const int* in_sizes, int n_in,
                              void* d_out, int out_size, void* d_ws, size_t ws_size,
                              hipStream_t stream) {
    const float* x            = (const float*)d_in[0];
    const int*   w_gate       = (const int*)d_in[1];
    const int*   w_up         = (const int*)d_in[2];
    const int*   w_down       = (const int*)d_in[3];
    const float* ws_gate      = (const float*)d_in[4];
    const float* ws_up        = (const float*)d_in[5];
    const float* ws_down      = (const float*)d_in[6];
    const float* b_gate       = (const float*)d_in[7];
    const float* b_up         = (const float*)d_in[8];
    const float* b_down       = (const float*)d_in[9];
    const float* a_scale_gate = (const float*)d_in[10];
    const float* a_scale_down = (const float*)d_in[11];
    float* out = (float*)d_out;

    signed char* qx  = (signed char*)d_ws;
    signed char* qwg = qx  + (size_t)T_TOK * H_DIM;
    signed char* qwu = qwg + (size_t)I_DIM * H_DIM;
    signed char* qwd = qwu + (size_t)I_DIM * H_DIM;
    signed char* qh  = qwd + (size_t)H_DIM * I_DIM;

    const int n4x = T_TOK * H_DIM / 4;
    quant_x_kernel<<<(n4x + 255) / 256, 256, 0, stream>>>(x, a_scale_gate, (int*)qx, n4x);

    const int n4w = I_DIM * H_DIM / 4;
    pack_w_kernel<<<(n4w + 255) / 256, 256, 0, stream>>>(w_gate, (int*)qwg, n4w);
    pack_w_kernel<<<(n4w + 255) / 256, 256, 0, stream>>>(w_up,   (int*)qwu, n4w);
    pack_w_kernel<<<(n4w + 255) / 256, 256, 0, stream>>>(w_down, (int*)qwd, n4w);

    gateup_kernel<<<dim3(I_DIM / 256, T_TOK / 64), 256, 0, stream>>>(
        qx, qwg, qwu, ws_gate, ws_up, b_gate, b_up, a_scale_gate, a_scale_down, qh);

    down_kernel<<<dim3(H_DIM / 256, T_TOK / 64), 256, 0, stream>>>(
        qh, qwd, ws_down, b_down, a_scale_down, out);
}
